// CausalRecurrentAttention_41532333752804
// MI455X (gfx1250) — matrix operand order repaired and checked by reading, not compiled, not run on hardware
//
#include <hip/hip_runtime.h>
#include <hip/hip_bf16.h>
#include <math.h>

// ---------------- problem constants ----------------
#define BV   2
#define TV   2048
#define CV   1024
#define HV   16
#define SV   16
#define HDV  64
#define NROW (BV * TV)   // 4096

typedef __attribute__((ext_vector_type(16))) __bf16 v16bf;
typedef __attribute__((ext_vector_type(8)))  __bf16 v8bf;
typedef __attribute__((ext_vector_type(8)))  float  v8f;

__device__ __forceinline__ float softplusf(float x) {
    return (x > 20.f) ? x : log1pf(__expf(x));
}

// ---------------- fp32 -> bf16 convert ----------------
__global__ void k_f32_to_bf16(const float* __restrict__ in, __bf16* __restrict__ out, int n) {
    int i = blockIdx.x * blockDim.x + threadIdx.x;
    if (i < n) out[i] = (__bf16)in[i];
}

// ---------------- W (RxC, f32) -> W^T (CxR, bf16) ----------------
__global__ void k_transpose_bf16(const float* __restrict__ W, __bf16* __restrict__ WT,
                                 int R, int Ccols) {
    __shared__ float tile[32][33];
    int c0 = blockIdx.x * 32, r0 = blockIdx.y * 32;
    int tx = threadIdx.x, ty = threadIdx.y;     // block (32,8)
#pragma unroll
    for (int i = 0; i < 32; i += 8)
        tile[ty + i][tx] = W[(size_t)(r0 + ty + i) * Ccols + (c0 + tx)];
    __syncthreads();
#pragma unroll
    for (int i = 0; i < 32; i += 8)
        WT[(size_t)(c0 + ty + i) * R + (r0 + tx)] = (__bf16)tile[tx][ty + i];
}

// ---------------- B_mat / C_mat: x @ WB , x @ WC (fp32, tiny) ----------------
__global__ void k_smallproj(const float* __restrict__ x, const float* __restrict__ WB,
                            const float* __restrict__ WC, float* __restrict__ Bm,
                            float* __restrict__ Cm) {
    int row = blockIdx.x;
    int tid = threadIdx.x;                      // 32 threads
    const float* xr = x + (size_t)row * CV;
    const float* W  = (tid < SV) ? WB : WC;
    int s = tid & (SV - 1);
    float acc = 0.f;
    for (int k = 0; k < CV; ++k) acc += xr[k] * W[k * SV + s];
    if (tid < SV) Bm[(size_t)row * SV + s] = acc;
    else          Cm[(size_t)row * SV + s] = acc;
}

// ---------------- sequential SSM scan ----------------
// 64-thread blocks, 32 blocks -> recurrence spread over many WGPs.
// B_t / C_t double-buffered in LDS via async global->LDS loads (ASYNCcnt),
// prefetched one timestep ahead to hide global latency in the serial loop.
__global__ void __launch_bounds__(64, 1)
k_scan(const float* __restrict__ xbase, const float* __restrict__ delta,
       const float* __restrict__ Bm, const float* __restrict__ Cm,
       const float* __restrict__ A_log, float* __restrict__ outfwd) {
    const int blocksPerB = CV / 64;             // 16
    const int b   = blockIdx.x / blocksPerB;
    const int c   = (blockIdx.x % blocksPerB) * 64 + threadIdx.x;
    const int tid = threadIdx.x;

    float Aneg[SV], h[SV];
#pragma unroll
    for (int s = 0; s < SV; ++s) {
        Aneg[s] = -__expf(A_log[c * SV + s]);
        h[s] = 0.f;
    }
    __shared__ alignas(16) float sBC[2][2 * SV];

    // prologue: async-stage B_0 / C_0 into buffer 0 (wave 0 lanes 0..31 only)
    if (tid < 2 * SV) {
        const float* g = (tid < SV) ? (Bm + (size_t)(b * TV) * SV + tid)
                                    : (Cm + (size_t)(b * TV) * SV + (tid - SV));
        unsigned ldsa = (unsigned)(size_t)&sBC[0][tid];
        unsigned long long ga = (unsigned long long)g;
        asm volatile("global_load_async_to_lds_b32 %0, %1, off" :: "v"(ldsa), "v"(ga) : "memory");
    }

    size_t base = (size_t)(b * TV) * CV + c;
    for (int t = 0; t < TV; ++t) {
        if (tid < 32) asm volatile("s_wait_asynccnt 0x0" ::: "memory");
        __syncthreads();                         // publish buffer t; prior reads of t+1's buffer done
        int buf = t & 1;
        if (tid < 2 * SV && (t + 1) < TV) {      // async-prefetch t+1 into the other buffer
            int rown = b * TV + t + 1;
            const float* g = (tid < SV) ? (Bm + (size_t)rown * SV + tid)
                                        : (Cm + (size_t)rown * SV + (tid - SV));
            unsigned ldsa = (unsigned)(size_t)&sBC[buf ^ 1][tid];
            unsigned long long ga = (unsigned long long)g;
            asm volatile("global_load_async_to_lds_b32 %0, %1, off" :: "v"(ldsa), "v"(ga) : "memory");
        }
        float d  = delta[base];
        float xb = xbase[base];
        __builtin_prefetch((const void*)(delta + base + CV), 0, 1);
        __builtin_prefetch((const void*)(xbase + base + CV), 0, 1);
        float dx = d * xb;
        const float* sB = sBC[buf];
        const float* sC = sBC[buf] + SV;
        float y = 0.f;
#pragma unroll
        for (int s = 0; s < SV; ++s) {
            float ba = __expf(d * Aneg[s]);
            h[s] = ba * h[s] + dx * sB[s];
            y += h[s] * sC[s];
        }
        outfwd[base] = y;
        base += CV;
    }
}

// ---------------- LayerNorm(x_base + out_fwd) -> bf16 ----------------
__global__ void k_layernorm(const float* __restrict__ xb, const float* __restrict__ of,
                            const float* __restrict__ g, const float* __restrict__ be,
                            __bf16* __restrict__ outB) {
    __shared__ float red[256];
    int row = blockIdx.x, tid = threadIdx.x;    // 256 threads
    const float* p1 = xb + (size_t)row * CV;
    const float* p2 = of + (size_t)row * CV;
    float v[4]; float s = 0.f;
#pragma unroll
    for (int i = 0; i < 4; ++i) { int idx = tid + i * 256; v[i] = p1[idx] + p2[idx]; s += v[i]; }
    red[tid] = s; __syncthreads();
    for (int w = 128; w > 0; w >>= 1) { if (tid < w) red[tid] += red[tid + w]; __syncthreads(); }
    float mu = red[0] * (1.f / CV);
    __syncthreads();
    float s2 = 0.f;
#pragma unroll
    for (int i = 0; i < 4; ++i) { float d = v[i] - mu; s2 += d * d; }
    red[tid] = s2; __syncthreads();
    for (int w = 128; w > 0; w >>= 1) { if (tid < w) red[tid] += red[tid + w]; __syncthreads(); }
    float rstd = rsqrtf(red[0] * (1.f / CV) + 1e-5f);
#pragma unroll
    for (int i = 0; i < 4; ++i) {
        int idx = tid + i * 256;
        outB[(size_t)row * CV + idx] = (__bf16)((v[i] - mu) * rstd * g[idx] + be[idx]);
    }
}

// ---------------- WMMA GEMM: A(MxK bf16) * Bt(NxK bf16)^T + bias --------------
// wave tile 32x64 (2x4 of 16x16), block = 4 waves => 64x128 per block
// MODE 0: f32 out             MODE 1: f32 softplus out
// MODE 2: bf16 out in (B,H,T,HD) scaled   MODE 3: bf16 out in (B,H,HD,T)
template <int MODE>
__global__ void __launch_bounds__(128, 1)
k_gemm(const __bf16* __restrict__ A, const __bf16* __restrict__ Bt,
       const float* __restrict__ bias, float* __restrict__ outF,
       __bf16* __restrict__ outB, int M, int N, int K, float scale) {
    const int lane = threadIdx.x & 31;
    const int wid  = threadIdx.x >> 5;
    const int laneLo = lane & 15, laneHi = lane >> 4;
    const int m0 = blockIdx.y * 64  + (wid & 1)  * 32;
    const int n0 = blockIdx.x * 128 + (wid >> 1) * 64;

    v8f acc[2][4];
#pragma unroll
    for (int i = 0; i < 2; ++i)
#pragma unroll
        for (int j = 0; j < 4; ++j)
#pragma unroll
            for (int e = 0; e < 8; ++e) acc[i][j][e] = 0.f;

    for (int k0 = 0; k0 < K; k0 += 32) {
        v16bf afrag[2], bfrag[4];
#pragma unroll
        for (int mt = 0; mt < 2; ++mt) {
            const __bf16* pa = A + (size_t)(m0 + mt * 16 + laneLo) * K + k0 + laneHi * 16;
            v8bf lo = *(const v8bf*)pa;
            v8bf hi = *(const v8bf*)(pa + 8);
#pragma unroll
            for (int e = 0; e < 8; ++e) { afrag[mt][e] = lo[e]; afrag[mt][e + 8] = hi[e]; }
            __builtin_prefetch((const void*)(pa + 32), 0, 1);   // next K tile
        }
#pragma unroll
        for (int nt = 0; nt < 4; ++nt)
            bfrag[nt] = *(const v16bf*)(Bt + (size_t)(n0 + nt * 16 + laneLo) * K + k0 + laneHi * 16);
#pragma unroll
        for (int mt = 0; mt < 2; ++mt)
#pragma unroll
            for (int nt = 0; nt < 4; ++nt)
                acc[mt][nt] = __builtin_amdgcn_wmma_f32_16x16x32_bf16(
                    false, afrag[mt], false, bfrag[nt], (short)0, acc[mt][nt], false, false);
    }

#pragma unroll
    for (int mt = 0; mt < 2; ++mt)
#pragma unroll
        for (int nt = 0; nt < 4; ++nt)
#pragma unroll
            for (int r = 0; r < 8; ++r) {
                int row = m0 + mt * 16 + laneHi * 8 + r;
                int col = n0 + nt * 16 + laneLo;
                float v = acc[mt][nt][r] + bias[col];
                if constexpr (MODE == 0) {
                    outF[(size_t)row * N + col] = v;
                } else if constexpr (MODE == 1) {
                    outF[(size_t)row * N + col] = softplusf(v);
                } else if constexpr (MODE == 2) {
                    int b = row >> 11, t = row & (TV - 1);
                    int h = col >> 6,  d = col & (HDV - 1);
                    outB[(((size_t)(b * HV + h)) * TV + t) * HDV + d] = (__bf16)(v * scale);
                } else {
                    int b = row >> 11, t = row & (TV - 1);
                    int h = col >> 6,  d = col & (HDV - 1);
                    outB[(((size_t)(b * HV + h)) * HDV + d) * TV + t] = (__bf16)v;
                }
            }
}

// ---------------- flash attention: wave = 16 queries, block = 4 waves = 64 queries ---
__global__ void __launch_bounds__(128, 1)
k_attn(const __bf16* __restrict__ Qb, const __bf16* __restrict__ Kb,
       const __bf16* __restrict__ VT, const float* __restrict__ temp,
       __bf16* __restrict__ outB) {
    const int lane = threadIdx.x & 31;
    const int wid  = threadIdx.x >> 5;
    const int laneLo = lane & 15, laneHi = lane >> 4;
    const int qblk = blockIdx.x & 31;
    const int bh   = blockIdx.x >> 5;
    const int qbase = qblk * 64 + wid * 16;
    const __bf16* Qp = Qb + (size_t)bh * TV * HDV;
    const __bf16* Kp = Kb + (size_t)bh * TV * HDV;
    const __bf16* Vp = VT + (size_t)bh * HDV * TV;
    const float tscale = log1pf(__expf(temp[0]));   // softplus(temp); 1/sqrt(HD) folded into Q

    __shared__ alignas(32) __bf16 sP[4][16 * 32];
    __bf16* myP = sP[wid];

    v16bf aQ[2];
    {
        const __bf16* pq = Qp + (size_t)(qbase + laneLo) * HDV;
#pragma unroll
        for (int kk = 0; kk < 2; ++kk) {
            v8bf lo = *(const v8bf*)(pq + kk * 32 + laneHi * 16);
            v8bf hi = *(const v8bf*)(pq + kk * 32 + 8 + laneHi * 16);
#pragma unroll
            for (int e = 0; e < 8; ++e) { aQ[kk][e] = lo[e]; aQ[kk][e + 8] = hi[e]; }
        }
    }

    v8f o[4];
#pragma unroll
    for (int nt = 0; nt < 4; ++nt)
#pragma unroll
        for (int e = 0; e < 8; ++e) o[nt][e] = 0.f;
    float mrow[8], lrow[8];
#pragma unroll
    for (int r = 0; r < 8; ++r) { mrow[r] = -3.0e38f; lrow[r] = 0.f; }

    const int kend = qbase + 16;
#pragma unroll 1
    for (int kc = 0; kc < kend; kc += 32) {
        v8f s[2];
#pragma unroll
        for (int nt = 0; nt < 2; ++nt)
#pragma unroll
            for (int e = 0; e < 8; ++e) s[nt][e] = 0.f;
        // scores = Q Kt
#pragma unroll
        for (int nt = 0; nt < 2; ++nt) {
            int key = kc + nt * 16 + laneLo;
#pragma unroll
            for (int kk = 0; kk < 2; ++kk) {
                v16bf bK = *(const v16bf*)(Kp + (size_t)key * HDV + kk * 32 + laneHi * 16);
                s[nt] = __builtin_amdgcn_wmma_f32_16x16x32_bf16(
                    false, aQ[kk], false, bK, (short)0, s[nt], false, false);
            }
        }
        // temperature + causal mask
#pragma unroll
        for (int nt = 0; nt < 2; ++nt) {
            int key = kc + nt * 16 + laneLo;
#pragma unroll
            for (int r = 0; r < 8; ++r) {
                int q = qbase + laneHi * 8 + r;
                float v = s[nt][r] * tscale;
                s[nt][r] = (key > q) ? -3.0e38f : v;
            }
        }
        // online softmax stats + O rescale (butterfly across the 16-lane half)
#pragma unroll
        for (int r = 0; r < 8; ++r) {
            float mx = fmaxf(s[0][r], s[1][r]);
#pragma unroll
            for (int off = 8; off > 0; off >>= 1) mx = fmaxf(mx, __shfl_xor(mx, off, 32));
            float mn = fmaxf(mrow[r], mx);
            float alpha = __expf(mrow[r] - mn);
            mrow[r] = mn;
            float p0 = __expf(s[0][r] - mn);
            float p1 = __expf(s[1][r] - mn);
            s[0][r] = p0; s[1][r] = p1;
            float rs = p0 + p1;
#pragma unroll
            for (int off = 8; off > 0; off >>= 1) rs += __shfl_xor(rs, off, 32);
            lrow[r] = lrow[r] * alpha + rs;
            o[0][r] *= alpha; o[1][r] *= alpha; o[2][r] *= alpha; o[3][r] *= alpha;
        }
        // stage P: C-layout -> LDS row-major 16x32, re-read as A fragment (per-wave region)
#pragma unroll
        for (int nt = 0; nt < 2; ++nt)
#pragma unroll
            for (int r = 0; r < 8; ++r)
                myP[(laneHi * 8 + r) * 32 + nt * 16 + laneLo] = (__bf16)s[nt][r];
        asm volatile("s_wait_dscnt 0x0" ::: "memory");
        v16bf aP;
        {
            v8bf lo = *(const v8bf*)(myP + laneLo * 32 + laneHi * 16);
            v8bf hi = *(const v8bf*)(myP + laneLo * 32 + 8 + laneHi * 16);
#pragma unroll
            for (int e = 0; e < 8; ++e) { aP[e] = lo[e]; aP[e + 8] = hi[e]; }
        }
        // accumulate O += P V
#pragma unroll
        for (int ntd = 0; ntd < 4; ++ntd) {
            int d = ntd * 16 + laneLo;
            v16bf bV = *(const v16bf*)(Vp + (size_t)d * TV + kc + laneHi * 16);
            o[ntd] = __builtin_amdgcn_wmma_f32_16x16x32_bf16(
                false, aP, false, bV, (short)0, o[ntd], false, false);
        }
    }
    // write (B,T,C) bf16 for final Wo GEMM
    int b = bh >> 4, h = bh & 15;
#pragma unroll
    for (int nt = 0; nt < 4; ++nt)
#pragma unroll
        for (int r = 0; r < 8; ++r) {
            int t = qbase + laneHi * 8 + r;
            int col = h * HDV + nt * 16 + laneLo;
            outB[((size_t)(b * TV + t)) * CV + col] = (__bf16)(o[nt][r] / lrow[r]);
        }
}

// ---------------- host side ----------------
extern "C" void kernel_launch(void* const* d_in, const int* in_sizes, int n_in,
                              void* d_out, int out_size, void* d_ws, size_t ws_size,
                              hipStream_t stream) {
    (void)in_sizes; (void)n_in; (void)out_size; (void)ws_size;
    const float* x     = (const float*)d_in[0];
    const float* A_log = (const float*)d_in[1];
    const float* Wd    = (const float*)d_in[2];
    const float* bd    = (const float*)d_in[3];
    const float* WB    = (const float*)d_in[4];
    const float* WC    = (const float*)d_in[5];
    const float* Wq    = (const float*)d_in[6];
    const float* bq    = (const float*)d_in[7];
    const float* Wk    = (const float*)d_in[8];
    const float* bk    = (const float*)d_in[9];
    const float* Wv    = (const float*)d_in[10];
    const float* bv    = (const float*)d_in[11];
    const float* Wx    = (const float*)d_in[12];
    const float* bx    = (const float*)d_in[13];
    const float* Wo    = (const float*)d_in[14];
    const float* bo    = (const float*)d_in[15];
    const float* ln_g  = (const float*)d_in[16];
    const float* ln_b  = (const float*)d_in[17];
    const float* temp  = (const float*)d_in[18];
    float* out = (float*)d_out;

    char* ws = (char*)d_ws;
    size_t off = 0;
    auto carve = [&](size_t bytes) -> char* {
        char* p = ws + off;
        off += (bytes + 255) & ~(size_t)255;
        return p;
    };
    __bf16* xbf    = (__bf16*)carve((size_t)NROW * CV * 2);
    __bf16* WxT    = (__bf16*)carve((size_t)CV * CV * 2);
    __bf16* WdT    = (__bf16*)carve((size_t)CV * CV * 2);
    __bf16* WqT    = (__bf16*)carve((size_t)CV * CV * 2);
    __bf16* WkT    = (__bf16*)carve((size_t)CV * CV * 2);
    __bf16* WvT    = (__bf16*)carve((size_t)CV * CV * 2);
    __bf16* WoT    = (__bf16*)carve((size_t)CV * CV * 2);
    float*  xbase  = (float*)carve((size_t)NROW * CV * 4);
    float*  delta  = (float*)carve((size_t)NROW * CV * 4);
    float*  outfwd = (float*)carve((size_t)NROW * CV * 4);
    float*  Bm     = (float*)carve((size_t)NROW * SV * 4);
    float*  Cm     = (float*)carve((size_t)NROW * SV * 4);
    __bf16* hybbf  = (__bf16*)carve((size_t)NROW * CV * 2);
    // reuse dead buffers: delta -> Q|K (bf16), outfwd -> V^T|attn (bf16)
    __bf16* Qbf    = (__bf16*)delta;
    __bf16* Kbf    = (__bf16*)((char*)delta + (size_t)NROW * CV * 2);
    __bf16* VTbf   = (__bf16*)outfwd;
    __bf16* attnbf = (__bf16*)((char*)outfwd + (size_t)NROW * CV * 2);

    // 1. convert activations to bf16
    int nx = NROW * CV;
    k_f32_to_bf16<<<(nx + 255) / 256, 256, 0, stream>>>(x, xbf, nx);
    // 2. transpose weights to NxK bf16
    dim3 tb(32, 8), tg(CV / 32, CV / 32);
    k_transpose_bf16<<<tg, tb, 0, stream>>>(Wx, WxT, CV, CV);
    k_transpose_bf16<<<tg, tb, 0, stream>>>(Wd, WdT, CV, CV);
    k_transpose_bf16<<<tg, tb, 0, stream>>>(Wq, WqT, CV, CV);
    k_transpose_bf16<<<tg, tb, 0, stream>>>(Wk, WkT, CV, CV);
    k_transpose_bf16<<<tg, tb, 0, stream>>>(Wv, WvT, CV, CV);
    k_transpose_bf16<<<tg, tb, 0, stream>>>(Wo, WoT, CV, CV);
    // 3. input projections
    dim3 gg(CV / 128, NROW / 64);
    k_gemm<0><<<gg, 128, 0, stream>>>(xbf, WxT, bx, xbase, nullptr, NROW, CV, CV, 1.f);
    k_gemm<1><<<gg, 128, 0, stream>>>(xbf, WdT, bd, delta, nullptr, NROW, CV, CV, 1.f);
    k_smallproj<<<NROW, 32, 0, stream>>>(x, WB, WC, Bm, Cm);
    // 4. sequential SSM scan (async B/C staging, 32 WGPs)
    k_scan<<<BV * (CV / 64), 64, 0, stream>>>(xbase, delta, Bm, Cm, A_log, outfwd);
    // 5. layernorm -> bf16 hybrid
    k_layernorm<<<NROW, 256, 0, stream>>>(xbase, outfwd, ln_g, ln_b, hybbf);
    // 6. QKV projections (Q pre-scaled by 1/sqrt(HD))
    k_gemm<2><<<gg, 128, 0, stream>>>(hybbf, WqT, bq, nullptr, Qbf, NROW, CV, CV, 0.125f);
    k_gemm<2><<<gg, 128, 0, stream>>>(hybbf, WkT, bk, nullptr, Kbf, NROW, CV, CV, 1.f);
    k_gemm<3><<<gg, 128, 0, stream>>>(hybbf, WvT, bv, nullptr, VTbf, NROW, CV, CV, 1.f);
    // 7. causal flash attention
    k_attn<<<BV * HV * (TV / 64), 128, 0, stream>>>(Qbf, Kbf, VTbf, temp, attnbf);
    // 8. output projection -> fp32 d_out
    k_gemm<0><<<gg, 128, 0, stream>>>(attnbf, WoT, bo, out, nullptr, NROW, CV, CV, 1.f);
}
